// BasicBlock_1w1a_7937099563427
// MI455X (gfx1250) — 8-bit matrix operand order repaired and checked by reading, compiled, not run on hardware
//
#include <hip/hip_runtime.h>
#include <hip/hip_bf16.h>
#include <stdint.h>


typedef __attribute__((ext_vector_type(8))) int v8i;

#define BATCH 64
#define CH    256
#define HH    32
#define WW    32
#define KTAPS 9
#define EPSV  1e-5f

__global__ __launch_bounds__(256) void pack_signs_nchw_to_nhwc(
    const float* __restrict__ in, signed char* __restrict__ out) {
  unsigned idx = blockIdx.x * 256u + threadIdx.x;
  unsigned c4   = idx & 63u;
  unsigned rest = idx >> 6;
  unsigned w    = rest & 31u;  rest >>= 5;
  unsigned h    = rest & 31u;
  unsigned b    = rest >> 5;
  if (b >= BATCH) return;

  uint32_t packed = 0;
#pragma unroll
  for (int j = 0; j < 4; ++j) {
    unsigned c = c4 * 4 + j;
    float v = in[((b * CH + c) * HH + h) * WW + w];
    int s = (v > 0.f) - (v < 0.f);
    packed |= ((uint32_t)(uint8_t)(int8_t)s) << (8 * j);
  }
  uint32_t* o32 = (uint32_t*)(out + (((size_t)b * HH + h) * WW + w) * CH + c4 * 4);
  *o32 = packed;
}

__global__ __launch_bounds__(256) void pack_sign_weights(
    const float* __restrict__ w, signed char* __restrict__ out) {
  unsigned idx = blockIdx.x * 256u + threadIdx.x;
  unsigned i4   = idx & 63u;
  unsigned rest = idx >> 6;
  unsigned o    = rest & 255u;
  unsigned t    = rest >> 8;
  if (t >= KTAPS) return;

  uint32_t packed = 0;
#pragma unroll
  for (int j = 0; j < 4; ++j) {
    unsigned i = i4 * 4 + j;
    float v = w[((size_t)o * CH + i) * KTAPS + t];
    int s = (v > 0.f) - (v < 0.f);
    packed |= ((uint32_t)(uint8_t)(int8_t)s) << (8 * j);
  }
  uint32_t* o32 = (uint32_t*)(out + (((size_t)t * CH + o) * CH) + i4 * 4);
  *o32 = packed;
}

__global__ __launch_bounds__(256) void binconv3x3_wmma_iu8(
    const signed char* __restrict__ Asig,
    const signed char* __restrict__ Wsig,
    const float* __restrict__ bias,
    const float* __restrict__ alpha,
    const float* __restrict__ bn_gamma,
    const float* __restrict__ bn_beta,
    const float* __restrict__ bn_mean,
    const float* __restrict__ bn_var,
    const float* __restrict__ residual,
    float* __restrict__ out,
    float stdw) {
  __shared__ __attribute__((aligned(16))) signed char ldsA[6 * 34 * 64];

  const int tid   = threadIdx.x;
  const int lane  = tid & 31;
  const int wave  = tid >> 5;
  const int waveM = wave & 3;
  const int waveN = wave >> 2;

  const int b  = blockIdx.x >> 3;
  const int h0 = (blockIdx.x & 7) * 4;
  const int n0 = blockIdx.y * 64 + waveN * 32 + (lane & 15);
  const int n1 = n0 + 16;

  v8i acc00 = {0,0,0,0,0,0,0,0};
  v8i acc01 = {0,0,0,0,0,0,0,0};
  v8i acc10 = {0,0,0,0,0,0,0,0};
  v8i acc11 = {0,0,0,0,0,0,0,0};

  const int wc0 = lane & 15;
  const int wc1 = wc0 + 16;
  const int kb2 = lane >> 4;
  const int kb16 = (lane >> 4) * 16;

  for (int cb = 0; cb < 4; ++cb) {
    __syncthreads();
    for (int idx = tid; idx < 816; idx += 256) {
      int pos = idx >> 2;
      int q   = idx & 3;
      int lr  = pos / 34;
      int lc  = pos - lr * 34;
      int h   = h0 - 1 + lr;
      int w   = lc - 1;
      int4 val = {0, 0, 0, 0};
      if ((unsigned)h < HH && (unsigned)w < WW) {
        val = *(const int4*)(Asig + (((size_t)b * HH + h) * WW + w) * CH +
                             cb * 64 + q * 16);
      }
      *(int4*)(ldsA + pos * 64 + q * 16) = val;
    }
    __syncthreads();

#pragma unroll
    for (int t = 0; t < KTAPS; ++t) {
      const int kh = t / 3;
      const int kw = t - kh * 3;
      const int ldsRow = (waveM + kh) * 34;

      const int2* ap0 = (const int2*)(ldsA + (ldsRow + wc0 + kw) * 64);
      const int2* ap1 = (const int2*)(ldsA + (ldsRow + wc1 + kw) * 64);
      int2 a0 = ap0[2 * kb2 + 0], a1 = ap0[2 * kb2 + 1], a2 = ap0[2 * kb2 + 4], a3 = ap0[2 * kb2 + 5];
      int2 c0 = ap1[2 * kb2 + 0], c1 = ap1[2 * kb2 + 1], c2 = ap1[2 * kb2 + 4], c3 = ap1[2 * kb2 + 5];
      v8i aF0, aF1;
      aF0[0]=a0.x; aF0[1]=a0.y; aF0[2]=a1.x; aF0[3]=a1.y;
      aF0[4]=a2.x; aF0[5]=a2.y; aF0[6]=a3.x; aF0[7]=a3.y;
      aF1[0]=c0.x; aF1[1]=c0.y; aF1[2]=c1.x; aF1[3]=c1.y;
      aF1[4]=c2.x; aF1[5]=c2.y; aF1[6]=c3.x; aF1[7]=c3.y;

      const int4* bp0 = (const int4*)(Wsig + (((size_t)t * CH + n0) * CH) +
                                      cb * 64 + kb16);
      const int4* bp1 = (const int4*)(Wsig + (((size_t)t * CH + n1) * CH) +
                                      cb * 64 + kb16);
      int4 b0 = bp0[0], b1 = bp0[2];
      int4 d0 = bp1[0], d1 = bp1[2];
      v8i bF0, bF1;
      bF0[0]=b0.x; bF0[1]=b0.y; bF0[2]=b0.z; bF0[3]=b0.w;
      bF0[4]=b1.x; bF0[5]=b1.y; bF0[6]=b1.z; bF0[7]=b1.w;
      bF1[0]=d0.x; bF1[1]=d0.y; bF1[2]=d0.z; bF1[3]=d0.w;
      bF1[4]=d1.x; bF1[5]=d1.y; bF1[6]=d1.z; bF1[7]=d1.w;

      acc00 = __builtin_amdgcn_wmma_i32_16x16x64_iu8(true, aF0, true, bF0, acc00, false, false);
      acc01 = __builtin_amdgcn_wmma_i32_16x16x64_iu8(true, aF0, true, bF1, acc01, false, false);
      acc10 = __builtin_amdgcn_wmma_i32_16x16x64_iu8(true, aF1, true, bF0, acc10, false, false);
      acc11 = __builtin_amdgcn_wmma_i32_16x16x64_iu8(true, aF1, true, bF1, acc11, false, false);
    }
  }

  const int hrow = h0 + waveM;
  const int mHi  = (lane >> 4) * 8;

  const float inv_0 = bn_gamma[n0] * rsqrtf(bn_var[n0] + EPSV);
  const float mu_0  = bn_mean[n0], bb_0 = bn_beta[n0];
  const float bi_0  = bias[n0],    al_0 = alpha[n0];
  const float inv_1 = bn_gamma[n1] * rsqrtf(bn_var[n1] + EPSV);
  const float mu_1  = bn_mean[n1], bb_1 = bn_beta[n1];
  const float bi_1  = bias[n1],    al_1 = alpha[n1];

  const size_t rowBase = ((size_t)b * CH) * HH + hrow;

#pragma unroll
  for (int r = 0; r < 8; ++r) {
    const int w0 = r + mHi;
    const int w1 = 16 + r + mHi;
    {
      size_t i00 = (((size_t)b * CH + n0) * HH + hrow) * WW + w0;
      float v = stdw * (float)acc00[r] + bi_0;
      v = v > 0.f ? v : al_0 * v;
      v = (v - mu_0) * inv_0 + bb_0;
      out[i00] = v + residual[i00];
    }
    {
      size_t i01 = (((size_t)b * CH + n1) * HH + hrow) * WW + w0;
      float v = stdw * (float)acc01[r] + bi_1;
      v = v > 0.f ? v : al_1 * v;
      v = (v - mu_1) * inv_1 + bb_1;
      out[i01] = v + residual[i01];
    }
    {
      size_t i10 = (((size_t)b * CH + n0) * HH + hrow) * WW + w1;
      float v = stdw * (float)acc10[r] + bi_0;
      v = v > 0.f ? v : al_0 * v;
      v = (v - mu_0) * inv_0 + bb_0;
      out[i10] = v + residual[i10];
    }
    {
      size_t i11 = (((size_t)b * CH + n1) * HH + hrow) * WW + w1;
      float v = stdw * (float)acc11[r] + bi_1;
      v = v > 0.f ? v : al_1 * v;
      v = (v - mu_1) * inv_1 + bb_1;
      out[i11] = v + residual[i11];
    }
  }
  (void)rowBase;
}

extern "C" void kernel_launch(void* const* d_in, const int* in_sizes, int n_in,
                              void* d_out, int out_size, void* d_ws, size_t ws_size,
                              hipStream_t stream) {
  const float* x      = (const float*)d_in[0];
  const float* W1     = (const float*)d_in[1];
  const float* b1     = (const float*)d_in[2];
  const float* W2     = (const float*)d_in[3];
  const float* b2     = (const float*)d_in[4];
  const float* alpha  = (const float*)d_in[5];
  const float* g1     = (const float*)d_in[6];
  const float* be1    = (const float*)d_in[7];
  const float* m1     = (const float*)d_in[8];
  const float* v1     = (const float*)d_in[9];
  const float* g2     = (const float*)d_in[10];
  const float* be2    = (const float*)d_in[11];
  const float* m2     = (const float*)d_in[12];
  const float* v2     = (const float*)d_in[13];
  float* out = (float*)d_out;

  signed char* wsA  = (signed char*)d_ws;
  signed char* wsW1 = wsA + (size_t)BATCH * HH * WW * CH;
  signed char* wsW2 = wsW1 + (size_t)KTAPS * CH * CH;

  const float stdw = 0.02946278254943948f;

  dim3 blk(256);
  dim3 gridPackX((BATCH * HH * WW * (CH / 4)) / 256);
  dim3 gridPackW((KTAPS * CH * (CH / 4) + 255) / 256);
  dim3 gridConv(BATCH * (HH / 4), CH / 64);

  pack_sign_weights<<<gridPackW, blk, 0, stream>>>(W1, wsW1);
  pack_sign_weights<<<gridPackW, blk, 0, stream>>>(W2, wsW2);
  pack_signs_nchw_to_nhwc<<<gridPackX, blk, 0, stream>>>(x, wsA);

  binconv3x3_wmma_iu8<<<gridConv, blk, 0, stream>>>(
      wsA, wsW1, b1, alpha, g1, be1, m1, v1, x, out, stdw);

  pack_signs_nchw_to_nhwc<<<gridPackX, blk, 0, stream>>>(out, wsA);

  binconv3x3_wmma_iu8<<<gridConv, blk, 0, stream>>>(
      wsA, wsW2, b2, alpha, g2, be2, m2, v2, out, out, stdw);
}
